// SelfAttention_29978871726340
// MI455X (gfx1250) — compile-verified, matrix operand order checked by reading, not run on hardware
//
#include <hip/hip_runtime.h>
#include <hip/hip_bf16.h>

// ---------------------------------------------------------------------------
// CDNA5 / gfx1250 fused attention block:  QKV proj (+RMSNorm+RoPE) ->
// causal GQA flash attention -> output proj.
// GEMMs: V_WMMA_F32_16X16X32_BF16 (wave32), fp32 accumulate.
// Global->LDS staging: Tensor Data Mover (tensor_load_to_lds, TENSORcnt),
// double-buffered in the projection GEMMs.
// ---------------------------------------------------------------------------

typedef __bf16 bf16_t;
typedef bf16_t v16bf __attribute__((ext_vector_type(16)));
typedef float  v8f   __attribute__((ext_vector_type(8)));
typedef unsigned int v4u __attribute__((ext_vector_type(4)));
typedef int v4i __attribute__((ext_vector_type(4)));
typedef int v8i __attribute__((ext_vector_type(8)));

static_assert(sizeof(v16bf) == 32, "v16bf must be 32 bytes");

union Frag {            // one WMMA A/B operand: 16 bf16 = two 16B chunks
    v16bf v;
    uint4 q[2];
    unsigned short u[16];
};

__device__ __forceinline__ unsigned short f2bf(float f) {
    unsigned int b = __float_as_uint(f);
    unsigned int r = b + 0x7FFFu + ((b >> 16) & 1u);   // round-to-nearest-even
    return (unsigned short)(r >> 16);
}

// ---------------------------------------------------------------------------
// Tensor Data Mover: 2D tile load, bf16 elements, LDS row padding +8 elements.
//   lds_addr     : LDS byte offset of tile destination
//   gaddr        : global byte address of tile start
//   tile_d0/d1   : tile width (elements, row) / height (rows)
//   tensor_d0    : tensor dim0 extent (elements)  (OOB clamp only; tiles fit)
//   stride0      : row stride (elements)
//   pad_int_code : pad_interval code (4 -> every 32 DWORDs, 5 -> every 64)
// pad_amount code 3 -> 4 DWORDs = 8 bf16 of padding per row.
// ---------------------------------------------------------------------------
__device__ __forceinline__ void tdm_load_2d(unsigned lds_addr, const void* gaddr,
                                            unsigned tile_d0, unsigned tile_d1,
                                            unsigned tensor_d0,
                                            unsigned long long stride0,
                                            unsigned pad_int_code)
{
    const unsigned long long ga = (unsigned long long)(uintptr_t)gaddr;
    const unsigned tensor_d1 = 1u << 20;   // generous extent, tiles never OOB

    v4u g0;
    g0[0] = 1u;                                          // count=1, user mode
    g0[1] = lds_addr;                                    // D#.lds_addr
    g0[2] = (unsigned)(ga & 0xFFFFFFFFu);                // global_addr[31:0]
    g0[3] = (unsigned)((ga >> 32) & 0x1FFFFFFu)          // global_addr[56:32]
          | (2u << 30);                                  // type = 2 ("image")

    v8i g1;
    g1[0] = (int)((1u << 16)                             // data_size = 2 bytes
                | (1u << 20)                             // pad_enable
                | (pad_int_code << 22)                   // pad_interval
                | (3u << 25));                           // pad_amount = 4 DW
    g1[1] = (int)((tensor_d0 & 0xFFFFu) << 16);          // tensor_dim0[15:0]
    g1[2] = (int)(((tensor_d0 >> 16) & 0xFFFFu)          // tensor_dim0[31:16]
                | ((tensor_d1 & 0xFFFFu) << 16));        // tensor_dim1[15:0]
    g1[3] = (int)(((tensor_d1 >> 16) & 0xFFFFu)          // tensor_dim1[31:16]
                | ((tile_d0 & 0xFFFFu) << 16));          // tile_dim0
    g1[4] = (int)(tile_d1 & 0xFFFFu);                    // tile_dim1 (tile_dim2=0)
    g1[5] = (int)(stride0 & 0xFFFFFFFFu);                // tensor_dim0_stride[31:0]
    g1[6] = (int)((stride0 >> 32) & 0xFFFFu);            // stride[47:32]
    g1[7] = 0;

    const v4i z4 = {0, 0, 0, 0};                         // groups 2/3 unused (2D)
    const v8i z8 = {0, 0, 0, 0, 0, 0, 0, 0};
    __builtin_amdgcn_tensor_load_to_lds(g0, g1, z4, z4, z8, 0);
}

// ---------------------------------------------------------------------------
// Problem constants
// ---------------------------------------------------------------------------
#define BATCH 2
#define SEQ   2048
#define HID   4096
#define NQH   32
#define NKVH  8
#define HD    128
#define MTOT  (BATCH * SEQ)     // 4096

// GEMM tiling: 256 threads = 8 waves; block tile 128(M) x 128(N); K-slab 64.
#define BM 128
#define BN 128
#define BK 64
#define AQ4 9          // (64+8) bf16 per padded LDS row = 9 uint4
#define CPAD 132       // fp32 staging row stride

union GemmSmem {
    struct { uint4 a[2][BM * AQ4]; uint4 b[2][BN * AQ4]; } ld;  // double-buffered
    struct { float c[BM * CPAD]; float rstd[BM]; } st;
};

__device__ __forceinline__ v8f wmma_bf16(const Frag& a, const Frag& b, v8f c) {
    return __builtin_amdgcn_wmma_f32_16x16x32_bf16(false, a.v, false, b.v,
                                                   (short)0, c, false, false);
}

// ---------------------------------------------------------------------------
// Core: C(128x128) = A(rows m0.., K=4096) * W(rows n0.., K=4096)^T
// A, W bf16 row-major, ld 4096 (K-contiguous NT GEMM).  TDM-fed, dual buffer.
// Result left in sm.st.c (fp32); barriers handled internally.
// ---------------------------------------------------------------------------
__device__ void gemm_core_nt(const unsigned short* __restrict__ A,
                             const unsigned short* __restrict__ W,
                             GemmSmem& sm)
{
    const int t    = threadIdx.x;
    const int lane = t & 31;
    const int wave = t >> 5;
    const int hs   = lane >> 4;   // half-wave select (K-interleave)
    const int ln   = lane & 15;

    v8f acc[8];
    #pragma unroll
    for (int n = 0; n < 8; n++)
        #pragma unroll
        for (int j = 0; j < 8; j++) acc[n][j] = 0.0f;

    const unsigned lds_a0 = (unsigned)(uintptr_t)&sm.ld.a[0][0];
    const unsigned lds_a1 = (unsigned)(uintptr_t)&sm.ld.a[1][0];
    const unsigned lds_b0 = (unsigned)(uintptr_t)&sm.ld.b[0][0];
    const unsigned lds_b1 = (unsigned)(uintptr_t)&sm.ld.b[1][0];

    // prologue: TDM-load slab 0 into buffer 0
    if (wave == 0) {
        tdm_load_2d(lds_a0, A, BK, BM, HID, HID, 4);
        tdm_load_2d(lds_b0, W, BK, BN, HID, HID, 4);
    }

    const int nslab = HID / BK;
    for (int i = 0; i < nslab; i++) {
        const int buf = i & 1;
        __syncthreads();   // compute of slab i-1 fully done everywhere
        if (wave == 0) {
            if (i + 1 < nslab) {   // issue slab i+1 into the other buffer
                const size_t koff = (size_t)(i + 1) * BK;
                tdm_load_2d(buf ? lds_a0 : lds_a1, A + koff, BK, BM, HID, HID, 4);
                tdm_load_2d(buf ? lds_b0 : lds_b1, W + koff, BK, BN, HID, HID, 4);
                __builtin_amdgcn_s_wait_tensorcnt(2);  // slab i's 2 loads retired
            } else {
                __builtin_amdgcn_s_wait_tensorcnt(0);
            }
        }
        __syncthreads();

        const uint4* sa = sm.ld.a[buf];
        const uint4* sb = sm.ld.b[buf];
        #pragma unroll
        for (int kk8 = 0; kk8 < BK / 8; kk8 += 4) {     // two K=32 sub-steps
            Frag fa;
            fa.q[0] = sa[(wave * 16 + ln) * AQ4 + kk8 + hs];
            fa.q[1] = sa[(wave * 16 + ln) * AQ4 + kk8 + 2 + hs];
            Frag fb[8];
            #pragma unroll
            for (int n = 0; n < 8; n++) {
                fb[n].q[0] = sb[(n * 16 + ln) * AQ4 + kk8 + hs];
                fb[n].q[1] = sb[(n * 16 + ln) * AQ4 + kk8 + 2 + hs];
            }
            #pragma unroll
            for (int n = 0; n < 8; n++)
                acc[n] = wmma_bf16(fa, fb[n], acc[n]);
        }
    }

    __syncthreads();   // LDS union: done reading ld.* before writing st.*
    #pragma unroll
    for (int n = 0; n < 8; n++)
        #pragma unroll
        for (int v = 0; v < 8; v++)
            sm.st.c[(wave * 16 + v + 8 * hs) * CPAD + n * 16 + ln] = acc[n][v];
    __syncthreads();
}

// ---------------------------------------------------------------------------
// Kernel: fp32 -> bf16 cast
// ---------------------------------------------------------------------------
__global__ void cast_f32_bf16(const float* __restrict__ src,
                              unsigned short* __restrict__ dst, int n)
{
    int i = blockIdx.x * blockDim.x + threadIdx.x;
    if (i < n) dst[i] = f2bf(src[i]);
}

// ---------------------------------------------------------------------------
// Kernel: fused QKV projection. gridDim = (32 m-tiles, 48 heads):
//   head 0..31  -> Q head (RMSNorm + RoPE), head-major bf16 out
//   head 32..39 -> K head (RMSNorm + RoPE)
//   head 40..47 -> V head (plain cast)
// ---------------------------------------------------------------------------
__global__ __launch_bounds__(256) void qkv_kernel(
    const unsigned short* __restrict__ xb,
    const unsigned short* __restrict__ wqb,
    const unsigned short* __restrict__ wkb,
    const unsigned short* __restrict__ wvb,
    const float* __restrict__ qnw, const float* __restrict__ knw,
    unsigned short* __restrict__ qh, unsigned short* __restrict__ kh,
    unsigned short* __restrict__ vh)
{
    __shared__ GemmSmem sm;
    const int m0   = blockIdx.x * BM;
    const int head = blockIdx.y;

    const unsigned short* W;
    int mode, hloc;
    if (head < 32)      { W = wqb + (size_t)head * HD * HID;        mode = 0; hloc = head; }
    else if (head < 40) { W = wkb + (size_t)(head - 32) * HD * HID; mode = 1; hloc = head - 32; }
    else                { W = wvb + (size_t)(head - 40) * HD * HID; mode = 2; hloc = head - 40; }

    gemm_core_nt(xb + (size_t)m0 * HID, W, sm);

    const int t = threadIdx.x;
    if (mode == 2) {   // V: straight bf16 store, head-major [b, hkv, s, d]
        for (int i = t; i < BM * HD; i += 256) {
            int row = i >> 7, col = i & 127;
            int gm = m0 + row, b = gm >> 11, s = gm & (SEQ - 1);
            vh[(((size_t)b * NKVH + hloc) * SEQ + s) * HD + col] = f2bf(sm.st.c[row * CPAD + col]);
        }
        return;
    }

    // per-row RMS over the 128-wide head (entirely inside this tile)
    if (t < BM) {
        float ss = 0.0f;
        for (int col = 0; col < HD; col++) {
            float v = sm.st.c[t * CPAD + col];
            ss += v * v;
        }
        sm.st.rstd[t] = rsqrtf(ss * (1.0f / 128.0f) + 1e-6f);
    }
    __syncthreads();

    const float* nw = (mode == 0) ? qnw : knw;
    unsigned short* dst = (mode == 0) ? qh : kh;
    const int nheads = (mode == 0) ? NQH : NKVH;
    const float LN1E6_64 = 13.815510558f / 64.0f;   // ln(1e6)/64

    for (int i = t; i < BM * HD; i += 256) {
        int row = i >> 7, col = i & 127;
        float r   = sm.st.rstd[row];
        float val = sm.st.c[row * CPAD + col] * r * nw[col];
        float par = sm.st.c[row * CPAD + (col ^ 64)] * r * nw[col ^ 64];
        int gm = m0 + row, b = gm >> 11, s = gm & (SEQ - 1);
        float invf = __expf(-(float)(col & 63) * LN1E6_64);
        float ang  = (float)s * invf;
        float c = cosf(ang), sn = sinf(ang);
        float out = (col < 64) ? (val * c - par * sn) : (val * c + par * sn);
        dst[(((size_t)b * nheads + hloc) * SEQ + s) * HD + col] = f2bf(out);
    }
}

// ---------------------------------------------------------------------------
// Kernel: causal GQA flash attention. gridDim = (16 q-tiles, 64 b*head).
// 8 waves x (16 q-rows x 128 d).  KV streamed in 32-row tiles.
// Q / K tiles fetched via TDM; V transposed into LDS by the waves.
// ---------------------------------------------------------------------------
#define QQ4 17   // (128+8) bf16 padded row = 17 uint4
#define VQ4 5    // (32+8)  bf16 padded row = 5 uint4

struct AttnSmem {
    uint4 q[128 * QQ4];   // Q tile, row-major [qrow][d]
    uint4 k[32 * QQ4];    // K tile, row-major [t][d]
    uint4 vt[128 * VQ4];  // V tile transposed [d][t]
    uint4 p[8][16 * VQ4]; // per-wave P staging (D-layout -> A-frag)
};

__global__ __launch_bounds__(256) void attn_kernel(
    const unsigned short* __restrict__ qh,
    const unsigned short* __restrict__ kh,
    const unsigned short* __restrict__ vh,
    unsigned short* __restrict__ ob)
{
    __shared__ AttnSmem sm;
    const int t    = threadIdx.x;
    const int lane = t & 31, wave = t >> 5;
    const int hs   = lane >> 4, ln = lane & 15;

    const int qs0 = blockIdx.x * 128;
    const int bh  = blockIdx.y;
    const int b   = bh >> 5, h = bh & 31;
    const int hkv = h >> 2;

    const unsigned short* qbase = qh + (((size_t)b * NQH + h) * SEQ + qs0) * HD;
    const unsigned short* kbase = kh + ((size_t)b * NKVH + hkv) * SEQ * HD;
    const unsigned short* vbase = vh + ((size_t)b * NKVH + hkv) * SEQ * HD;

    // --- Q tile via TDM (128x128, rows padded to 136) ---
    if (wave == 0) {
        tdm_load_2d((unsigned)(uintptr_t)&sm.q[0], qbase, HD, 128, HD, HD, 5);
        __builtin_amdgcn_s_wait_tensorcnt(0);
    }
    __syncthreads();

    Frag fq[4];
    #pragma unroll
    for (int d = 0; d < 4; d++) {
        fq[d].q[0] = sm.q[(wave * 16 + ln) * QQ4 + d * 4 + hs];
        fq[d].q[1] = sm.q[(wave * 16 + ln) * QQ4 + d * 4 + 2 + hs];
    }

    v8f   o[8];
    float mrow[8], lrow[8];
    #pragma unroll
    for (int n = 0; n < 8; n++)
        #pragma unroll
        for (int j = 0; j < 8; j++) o[n][j] = 0.0f;
    #pragma unroll
    for (int v = 0; v < 8; v++) { mrow[v] = -1e30f; lrow[v] = 0.0f; }

    const int   ntiles = (qs0 + 128) >> 5;           // causal: only past KV tiles
    const float scale  = 0.088388347648318447f;      // 1/sqrt(128)

    unsigned short* vts = reinterpret_cast<unsigned short*>(&sm.vt[0]);

    for (int kt = 0; kt < ntiles; kt++) {
        const int t0 = kt << 5;
        __syncthreads();
        // K tile via TDM (32x128 rows padded to 136)
        if (wave == 0)
            tdm_load_2d((unsigned)(uintptr_t)&sm.k[0],
                        kbase + (size_t)t0 * HD, HD, 32, HD, HD, 5);
        // V tile transposed on the way into LDS (manual: TDM can't transpose)
        for (int i = t; i < 32 * 16; i += 256) {
            int row = i >> 4, cc = (i & 15) * 8;
            uint4 vv = *reinterpret_cast<const uint4*>(vbase + (size_t)(t0 + row) * HD + cc);
            const unsigned short* pv = reinterpret_cast<const unsigned short*>(&vv);
            #pragma unroll
            for (int j = 0; j < 8; j++) vts[(cc + j) * 40 + row] = pv[j];
        }
        if (wave == 0) __builtin_amdgcn_s_wait_tensorcnt(0);
        __syncthreads();

        // ---- S = Q @ K^T  (16 x 32 per wave, two N-subtiles) ----
        v8f sacc[2];
        #pragma unroll
        for (int n = 0; n < 2; n++)
            #pragma unroll
            for (int j = 0; j < 8; j++) sacc[n][j] = 0.0f;
        #pragma unroll
        for (int d = 0; d < 4; d++) {
            Frag fk[2];
            #pragma unroll
            for (int n = 0; n < 2; n++) {
                fk[n].q[0] = sm.k[(n * 16 + ln) * QQ4 + d * 4 + hs];
                fk[n].q[1] = sm.k[(n * 16 + ln) * QQ4 + d * 4 + 2 + hs];
            }
            #pragma unroll
            for (int n = 0; n < 2; n++)
                sacc[n] = wmma_bf16(fq[d], fk[n], sacc[n]);
        }

        // ---- online softmax (per-row stats live in a 16-lane half-wave) ----
        float pnew[2][8];
        #pragma unroll
        for (int v = 0; v < 8; v++) {
            const int qrow = qs0 + wave * 16 + v + 8 * hs;
            float m_new = mrow[v];
            #pragma unroll
            for (int n = 0; n < 2; n++) {
                int tp = t0 + n * 16 + ln;
                float sv = sacc[n][v] * scale;
                if (tp > qrow) sv = -1e30f;          // causal mask
                sacc[n][v] = sv;
                m_new = fmaxf(m_new, sv);
            }
            #pragma unroll
            for (int off = 1; off < 16; off <<= 1)
                m_new = fmaxf(m_new, __shfl_xor(m_new, off, 32));
            const float corr = __expf(mrow[v] - m_new);
            float lsum = 0.0f;
            #pragma unroll
            for (int n = 0; n < 2; n++) {
                float p = __expf(sacc[n][v] - m_new);
                pnew[n][v] = p;
                lsum += p;
            }
            #pragma unroll
            for (int off = 1; off < 16; off <<= 1)
                lsum += __shfl_xor(lsum, off, 32);
            lrow[v] = lrow[v] * corr + lsum;
            mrow[v] = m_new;
            #pragma unroll
            for (int n = 0; n < 8; n++) o[n][v] *= corr;
        }

        // ---- relayout P (D-layout) -> A-frag via wave-private LDS ----
        unsigned short* pws = reinterpret_cast<unsigned short*>(&sm.p[wave][0]);
        #pragma unroll
        for (int n = 0; n < 2; n++)
            #pragma unroll
            for (int v = 0; v < 8; v++)
                pws[(v + 8 * hs) * 40 + n * 16 + ln] = f2bf(pnew[n][v]);
        // DS ops from one wave are in-order; gather back as A-frag:
        Frag fp;
        fp.q[0] = sm.p[wave][ln * VQ4 + hs];
        fp.q[1] = sm.p[wave][ln * VQ4 + 2 + hs];

        // ---- O += P @ V  (K = 32 kv positions) ----
        Frag fv[8];
        #pragma unroll
        for (int n = 0; n < 8; n++) {
            fv[n].q[0] = sm.vt[(n * 16 + ln) * VQ4 + hs];
            fv[n].q[1] = sm.vt[(n * 16 + ln) * VQ4 + 2 + hs];
        }
        #pragma unroll
        for (int n = 0; n < 8; n++)
            o[n] = wmma_bf16(fp, fv[n], o[n]);
    }

    // ---- finalize: O / l, store bf16 row-major [b*S+s][h*128+d] ----
    #pragma unroll
    for (int v = 0; v < 8; v++) {
        int row = wave * 16 + v + 8 * hs;
        int s = qs0 + row;
        float inv = 1.0f / lrow[v];
        size_t base = ((size_t)b * SEQ + s) * (NQH * HD) + (size_t)h * HD;
        #pragma unroll
        for (int n = 0; n < 8; n++)
            ob[base + n * 16 + ln] = f2bf(o[n][v] * inv);
    }
}

// ---------------------------------------------------------------------------
// Kernel: output projection  out = o @ wo^T  (fp32 result)
// ---------------------------------------------------------------------------
__global__ __launch_bounds__(256) void oproj_kernel(
    const unsigned short* __restrict__ obf,
    const unsigned short* __restrict__ wob,
    float* __restrict__ out)
{
    __shared__ GemmSmem sm;
    const int m0 = blockIdx.x * BM;
    const int n0 = blockIdx.y * BN;
    gemm_core_nt(obf + (size_t)m0 * HID, wob + (size_t)n0 * HID, sm);
    for (int i = threadIdx.x; i < BM * BN; i += 256) {
        int row = i >> 7, col = i & 127;
        out[(size_t)(m0 + row) * HID + n0 + col] = sm.st.c[row * CPAD + col];
    }
}

// ---------------------------------------------------------------------------
// Host launcher
// ---------------------------------------------------------------------------
extern "C" void kernel_launch(void* const* d_in, const int* in_sizes, int n_in,
                              void* d_out, int out_size, void* d_ws, size_t ws_size,
                              hipStream_t stream)
{
    (void)in_sizes; (void)n_in; (void)out_size; (void)ws_size;
    const float* x   = (const float*)d_in[0];
    const float* wq  = (const float*)d_in[1];
    const float* wk  = (const float*)d_in[2];
    const float* wv  = (const float*)d_in[3];
    const float* wo  = (const float*)d_in[4];
    const float* qnw = (const float*)d_in[5];
    const float* knw = (const float*)d_in[6];
    float* out = (float*)d_out;

    const size_t M = MTOT, Dq = NQH * HD, Dkv = NKVH * HD;
    char* ws = (char*)d_ws;
    unsigned short* xb  = (unsigned short*)ws; ws += M * HID * 2;
    unsigned short* wqb = (unsigned short*)ws; ws += Dq * HID * 2;
    unsigned short* wkb = (unsigned short*)ws; ws += Dkv * HID * 2;
    unsigned short* wvb = (unsigned short*)ws; ws += Dkv * HID * 2;
    unsigned short* wob = (unsigned short*)ws; ws += HID * Dq * 2;
    unsigned short* qhd = (unsigned short*)ws; ws += M * Dq * 2;
    unsigned short* khd = (unsigned short*)ws; ws += M * Dkv * 2;
    unsigned short* vhd = (unsigned short*)ws; ws += M * Dkv * 2;
    unsigned short* obf = (unsigned short*)ws; ws += M * Dq * 2;

    auto cast = [&](const float* s, unsigned short* d, size_t n) {
        cast_f32_bf16<<<dim3((unsigned)((n + 255) / 256)), 256, 0, stream>>>(s, d, (int)n);
    };
    cast(x,  xb,  M * HID);
    cast(wq, wqb, Dq * HID);
    cast(wk, wkb, Dkv * HID);
    cast(wv, wvb, Dkv * HID);
    cast(wo, wob, HID * Dq);

    qkv_kernel<<<dim3(MTOT / BM, 48), 256, 0, stream>>>(xb, wqb, wkb, wvb, qnw, knw,
                                                        qhd, khd, vhd);
    attn_kernel<<<dim3(SEQ / 128, BATCH * NQH), 256, 0, stream>>>(qhd, khd, vhd, obf);
    oproj_kernel<<<dim3(MTOT / BM, HID / BN), 256, 0, stream>>>(obf, wob, out);
}
